// GraphAttentionBias_12816182411320
// MI455X (gfx1250) — matrix operand order repaired and checked by reading, not compiled, not run on hardware
//
#include <hip/hip_runtime.h>
#include <hip/hip_bf16.h>

// Problem constants (match reference)
#define N_NODES 2048
#define E_EDGES 32768      // 2^15
#define B_BATCH 8
#define NUM_ET  16
#define EPS_F   1e-9f

typedef __attribute__((ext_vector_type(16))) _Float16 v16h;
typedef __attribute__((ext_vector_type(8)))  float    v8f;

// Fast SiLU: x * sigmoid(x) with v_exp_f32 + v_rcp_f32 (no IEEE div sequence).
__device__ __forceinline__ float silu_f(float x) {
    return x * __builtin_amdgcn_rcpf(1.0f + __expf(-x));
}

// ---------------------------------------------------------------------------
// Kernel 1: fill bias tensor with non_edge_bias. 134 MB of f32 -> pure
// store-bandwidth problem. float4 (b128) stores, regular temporal hint so the
// whole tensor stays resident in the 192 MB L2 for the scatter kernel.
// ---------------------------------------------------------------------------
__global__ void fill_bias_kernel(float* __restrict__ out,
                                 const float* __restrict__ non_edge_bias) {
    const float v = non_edge_bias[0];
    const size_t idx = (size_t)blockIdx.x * blockDim.x + threadIdx.x;
    float4 f4 = make_float4(v, v, v, v);
    reinterpret_cast<float4*>(out)[idx] = f4;
}

// ---------------------------------------------------------------------------
// Kernel 2: edge MLP via v_wmma_f32_16x16x32_f16 + symmetric scatter.
// One wave handles 16 (batch,edge) rows:
//   layer1: A(16x32 f16, features padded 10->32) x W1-tiles(32x16 f16) x2
//   layer2: SiLU + W2 dot folded into D-register layout + shfl_xor reduction
// ---------------------------------------------------------------------------
__global__ void __launch_bounds__(256)
edge_bias_kernel(const float* __restrict__ xyz,
                 const int*   __restrict__ edge_index,   // [2, E]
                 const int*   __restrict__ edge_type,    // [E]
                 const float* __restrict__ rest_len,     // [E]
                 const float* __restrict__ emb,          // [16, 8]
                 const float* __restrict__ W1,           // [10, 32] row-major
                 const float* __restrict__ b1,           // [32]
                 const float* __restrict__ W2,           // [32]
                 const float* __restrict__ b2,           // [1]
                 float*       __restrict__ out) {        // [B, N, N]
    const int lane = threadIdx.x & 31;
    const int wave = threadIdx.x >> 5;
    const int half = lane >> 4;       // 0: lanes 0-15, 1: lanes 16-31
    const int l16  = lane & 15;

    const int tile = blockIdx.x * (blockDim.x >> 5) + wave;
    const int row0 = tile << 4;       // 16 rows (= batch*E flat index) per tile

    // ---- per-lane edge features for row m = l16 (both half-waves need the
    //      same row's features, on disjoint K subsets) ----
    const int row = row0 + l16;            // row = b*E + e
    const int b   = row >> 15;             // E = 2^15
    const int e   = row & (E_EDGES - 1);
    const int vi  = edge_index[e];
    const int vj  = edge_index[E_EDGES + e];
    const float* pi = xyz + ((size_t)b * N_NODES + vi) * 3;
    const float* pj = xyz + ((size_t)b * N_NODES + vj) * 3;
    const float dx = pi[0] - pj[0];
    const float dy = pi[1] - pj[1];
    const float dz = pi[2] - pj[2];
    const float dist2 = dx * dx + dy * dy + dz * dz;
    const float dist  = sqrtf(dist2 + EPS_F);
    const float rest  = rest_len[e];
    const float delta = (dist - rest) * __builtin_amdgcn_rcpf(rest + EPS_F);
    const float* sp   = emb + edge_type[e] * 8;
    const float s0 = sp[0], s1 = sp[1], s2 = sp[2], s3 = sp[3];
    const float s4 = sp[4], s5 = sp[5], s6 = sp[6], s7 = sp[7];

    // A tile: 16x32 f16, rows = edges, K = features (10 real, rest zero).
    // ISA layout: lanes 0-15 hold M=l16, elements 0..7 -> K=0..7, 8..15 -> K=16..23
    //             lanes 16-31 hold M=l16, elements 0..7 -> K=8..15, 8..15 -> K=24..31
    // Only K<10 is nonzero: half 0 -> K=0..7 = {dist2,delta,s0..s5};
    //                       half 1 -> K=8,9  = {s6,s7}; everything else 0.
    v16h a;
    a[0] = (_Float16)(half ? s6 : dist2);
    a[1] = (_Float16)(half ? s7 : delta);
    a[2] = (_Float16)(half ? 0.0f : s0);
    a[3] = (_Float16)(half ? 0.0f : s1);
    a[4] = (_Float16)(half ? 0.0f : s2);
    a[5] = (_Float16)(half ? 0.0f : s3);
    a[6] = (_Float16)(half ? 0.0f : s4);
    a[7] = (_Float16)(half ? 0.0f : s5);
#pragma unroll
    for (int t = 8; t < 16; ++t) a[t] = (_Float16)0.0f;

    // B tiles: 32x16 f16 slices of W1 (cols 0-15 and 16-31).
    // ISA layout: lanes 0-15 hold N=l16, elements 0..7 -> K=0..7, 8..15 -> K=16..23;
    //             lanes 16-31 elements 0..7 -> K=8..15, 8..15 -> K=24..31 (K>=10 zero).
    v16h bw0, bw1;
#pragma unroll
    for (int t = 0; t < 16; ++t) {
        float w0 = 0.0f, w1 = 0.0f;
        if (t < 8 && half * 8 + t < 10) {
            float u0 = W1[(half * 8 + t) * 32 + l16];
            float u1 = W1[(half * 8 + t) * 32 + 16 + l16];
            w0 = u0;
            w1 = u1;
        }
        bw0[t] = (_Float16)w0;
        bw1[t] = (_Float16)w1;
    }

    // Layer 1: two 16x16x32 WMMAs, f32 accumulate from zero.
    v8f c0 = {};
    v8f c1 = {};
    c0 = __builtin_amdgcn_wmma_f32_16x16x32_f16(false, a, false, bw0,
                                                (short)0, c0, false, false);
    c1 = __builtin_amdgcn_wmma_f32_16x16x32_f16(false, a, false, bw1,
                                                (short)0, c1, false, false);

    // Layer 2 in D-register layout: VGPR r holds (M = r + 8*half, N = l16).
    // bias[m] = sum_n silu(h[m][n] + b1[n]) * W2[n]  -> elementwise weight +
    // 16-lane xor-reduction (masks 1,2,4,8 never cross the half-wave boundary).
    const float b1a = b1[l16];
    const float b1b = b1[16 + l16];
    const float w2a = W2[l16];
    const float w2b = W2[16 + l16];
    const float b2v = b2[0];

    float res[8];
#pragma unroll
    for (int r = 0; r < 8; ++r) {
        float h0 = silu_f(c0[r] + b1a);
        float h1 = silu_f(c1[r] + b1b);
        float acc = h0 * w2a + h1 * w2b;
        acc += __shfl_xor(acc, 1);
        acc += __shfl_xor(acc, 2);
        acc += __shfl_xor(acc, 4);
        acc += __shfl_xor(acc, 8);
        res[r] = acc + b2v;
    }

    // Scatter: lane 0 owns rows 0..7, lane 16 owns rows 8..15 (random-access
    // writes, mostly L2 hits since the fill kernel left the tensor resident).
    if (l16 == 0) {
#pragma unroll
        for (int r = 0; r < 8; ++r) {
            const int m  = half * 8 + r;
            const int rw = row0 + m;
            const int bb = rw >> 15;
            const int ee = rw & (E_EDGES - 1);
            const int ii = edge_index[ee];
            const int jj = edge_index[E_EDGES + ee];
            const size_t base = (size_t)bb * N_NODES * N_NODES;
            const float v = res[r];
            out[base + (size_t)ii * N_NODES + jj] = v;
            out[base + (size_t)jj * N_NODES + ii] = v;
        }
    }
}

// ---------------------------------------------------------------------------
// Launcher
// ---------------------------------------------------------------------------
extern "C" void kernel_launch(void* const* d_in, const int* in_sizes, int n_in,
                              void* d_out, int out_size, void* d_ws, size_t ws_size,
                              hipStream_t stream) {
    const float* xyz        = (const float*)d_in[0];
    const int*   edge_index = (const int*)  d_in[1];
    const int*   edge_type  = (const int*)  d_in[2];
    const float* rest_len   = (const float*)d_in[3];
    const float* emb        = (const float*)d_in[4];
    const float* W1         = (const float*)d_in[5];
    const float* b1         = (const float*)d_in[6];
    const float* W2         = (const float*)d_in[7];
    const float* b2         = (const float*)d_in[8];
    const float* neb        = (const float*)d_in[9];
    float*       out        = (float*)d_out;

    // Fill: B*N*N = 33,554,432 floats; float4 per thread -> 8,388,608 threads.
    const size_t total = (size_t)B_BATCH * N_NODES * N_NODES;
    dim3 fill_block(256);
    dim3 fill_grid((unsigned)((total / 4 + 255) / 256));
    fill_bias_kernel<<<fill_grid, fill_block, 0, stream>>>(out, neb);

    // Edge MLP + scatter: B*E/16 = 16,384 tiles, 8 waves (tiles) per block.
    const int tiles = (B_BATCH * E_EDGES) / 16;
    dim3 edge_block(256);
    dim3 edge_grid(tiles / 8);
    edge_bias_kernel<<<edge_grid, edge_block, 0, stream>>>(
        xyz, edge_index, edge_type, rest_len, emb, W1, b1, W2, b2, out);
}
